// SlidingWindowAttention_6107443495283
// MI455X (gfx1250) — matrix operand order repaired and checked by reading, not compiled, not run on hardware
//
#include <hip/hip_runtime.h>
#include <cstdint>

// ---------------------------------------------------------------------------
// Sliding-window attention for MI455X (gfx1250, wave32, WMMA + async-to-LDS).
//   B=2, T=2048, C=1024, H=16, D=64, WINDOW=64, ROPE_BASE=1e4
// Heavy GEMMs (qkv + proj, ~95% of FLOPs): LDS double-buffered WMMA GEMM with
// GLOBAL_LOAD_ASYNC_TO_LDS_B128 copies (ASYNCcnt) overlapping compute.
// Attention (<=65 keys/query, L2-resident): VALU online softmax + prefetch.
// Workspace layout (72 MB total):
//   [ 0MB) Xh   f16  4096x1024
//   [ 8MB) WAh  f16  3072x1024
//   [14MB) WPh  f16  1024x1024
//   [16MB) qkv  f32  4096x3072   (RoPE applied in place)
//   [64MB) CTXh f16  4096x1024
// ---------------------------------------------------------------------------

typedef _Float16 half_t;
typedef __attribute__((ext_vector_type(16))) _Float16 v16h;
typedef __attribute__((ext_vector_type(8)))  float    v8f;

union Frag16 {            // one lane's share of a 16x32 (A) or 32x16 (B) f16 tile
  v16h  v;
  uint4 w[2];
};

// ---------------------------------------------------------------- cast f32->f16
__global__ void cast_f32_to_f16(const float* __restrict__ src,
                                half_t* __restrict__ dst, int n) {
  int i = blockIdx.x * blockDim.x + threadIdx.x;
  if (i < n) dst[i] = (half_t)src[i];
}

// ------------------------------------------------------------------- WMMA GEMM
// C[M,N] (f32) = A[M,K] (f16 row-major) * B[N,K]^T (f16 row-major).
// grid = (M/64, N/64), block = 128 (4 waves). Block tile 64x64.
// k-step = 32. A/B 64x32 tiles staged in LDS (row stride 40 halves = 80B),
// double buffered, filled with global_load_async_to_lds_b128 (16B per lane,
// 4 issues per thread per step). Each wave computes a 16x64 strip: all 5
// fragments (1 A + 4 B) are ds_load'ed first, then 4 WMMAs issue with partial
// DScnt waits so LDS latency overlaps matrix work.
//
// CDNA5 fragment layouts (wave32):
//  A 16x32 f16 : lane l -> row = l&15 ; halves k = (l>>4)*8 + [0..7]
//                                     and k = 16 + (l>>4)*8 + [0..7]
//  B 32x16 f16 : lane l -> col = l&15 ; same k order as A
//  C 16x16 f32 : VGPR r -> (M = r + 8*(l>>4), N = l&15)
__global__ void __launch_bounds__(128)
wmma_gemm_nt(const half_t* __restrict__ A, const half_t* __restrict__ Bw,
             float* __restrict__ C, int M, int N, int K) {
  // [buf][A(64x40) | B(64x40)] halves : 2 * (2560 + 2560) = 10240 halves = 20KB
  __shared__ __align__(16) half_t smem[10240];

  const int tid  = threadIdx.x;
  const int lane = tid & 31;
  const int wave = tid >> 5;
  const int mblk = blockIdx.x * 64;
  const int nblk = blockIdx.y * 64;
  const int row  = lane & 15;
  const int hi   = lane >> 4;

  const half_t* Ag = A  + (size_t)mblk * K;
  const half_t* Bg = Bw + (size_t)nblk * K;
  const uint32_t lds_base = (uint32_t)(uintptr_t)smem;  // LDS byte address

  const int nk = K >> 5;

  // Issue this thread's share of the 64x32 A and B tile copies for (buf, kb):
  // 256 16B chunks per tile, 128 threads -> 2 chunks each per tile.
  auto issue_copies = [&](int buf, int kb) {
    uint32_t la = lds_base + (uint32_t)buf * 10240u;  // bytes
    uint32_t lb = la + 5120u;
#pragma unroll
    for (int s = 0; s < 2; ++s) {
      int c   = tid + (s << 7);
      int r   = c >> 2;                 // tile row 0..63
      int col = c & 3;                  // 16B chunk in row
      uint32_t lo = (uint32_t)(r * 80 + col * 16);
      const half_t* ga = Ag + (size_t)r * K + kb + col * 8;
      const half_t* gb = Bg + (size_t)r * K + kb + col * 8;
      asm volatile("global_load_async_to_lds_b128 %0, %1, off"
                   :: "v"(la + lo), "v"((unsigned long long)(uintptr_t)ga)
                   : "memory");
      asm volatile("global_load_async_to_lds_b128 %0, %1, off"
                   :: "v"(lb + lo), "v"((unsigned long long)(uintptr_t)gb)
                   : "memory");
    }
  };

  v8f acc[4] = {};
  issue_copies(0, 0);                       // prologue: fill buffer 0
  for (int i = 0; i < nk; ++i) {
    const int buf = i & 1;
    // Prefetch next tile into the other buffer (wrap to kb=0 on the last step
    // so ASYNCcnt bookkeeping stays uniform; s_endpgm wait-idles the tail).
    const int kbn = (i + 1 < nk) ? ((i + 1) << 5) : 0;
    issue_copies(buf ^ 1, kbn);
    // Wait until only the 4 just-issued copies are outstanding -> current
    // buffer's copies (issued one iteration ago, in-order completion) done.
#if __has_builtin(__builtin_amdgcn_s_wait_asynccnt)
    __builtin_amdgcn_s_wait_asynccnt(4);
#else
    asm volatile("s_wait_asynccnt 0x4" ::: "memory");
#endif
    __syncthreads();                        // all waves' copies visible

    const half_t* SA = smem + buf * 5120;   // halves
    const half_t* SB = SA + 2560;

    // Load ALL fragments first (one ds_load burst), then issue the WMMAs so
    // the compiler can use partial s_wait_dscnt values instead of 0 each time.
    Frag16 a;
    {
      const half_t* ap = SA + (wave * 16 + row) * 40;
      a.w[0] = *(const uint4*)(ap + hi * 8);
      a.w[1] = *(const uint4*)(ap + 16 + hi * 8);
    }
    Frag16 b[4];
#pragma unroll
    for (int t = 0; t < 4; ++t) {
      const half_t* bp = SB + (t * 16 + row) * 40;
      b[t].w[0] = *(const uint4*)(bp + hi * 8);
      b[t].w[1] = *(const uint4*)(bp + 16 + hi * 8);
    }
#pragma unroll
    for (int t = 0; t < 4; ++t) {
      acc[t] = __builtin_amdgcn_wmma_f32_16x16x32_f16(
          /*neg_a=*/false, a.v, /*neg_b=*/false, b[t].v,
          /*c_mod=*/(short)0, acc[t], /*reuse_a=*/false, /*reuse_b=*/false);
    }
    __syncthreads();                        // reads done before next overwrite
  }

  const int m0 = mblk + wave * 16;
#pragma unroll
  for (int t = 0; t < 4; ++t) {
    float* cp = C + (size_t)(m0 + hi * 8) * N + (nblk + t * 16 + row);
#pragma unroll
    for (int r = 0; r < 8; ++r) cp[(size_t)r * N] = acc[t][r];
  }
}

// ----------------------------------------------------------- RoPE (in place)
__global__ void rope_inplace(float* __restrict__ qkv, int Bb, int Tt, int C,
                             int H, int D) {
  const int half = D >> 1;
  int idx = blockIdx.x * blockDim.x + threadIdx.x;
  int total = Bb * Tt * H * half;
  if (idx >= total) return;
  int i = idx % half;
  int h = (idx / half) % H;
  int t = (idx / (half * H)) % Tt;
  int b = idx / (half * H * Tt);

  float inv = __expf(-((float)(2 * i) / (float)D) * __logf(10000.0f));
  float fr  = (float)t * inv;
  float c = __cosf(fr), s = __sinf(fr);

  size_t base = (size_t)(b * Tt + t) * (size_t)(3 * C);
  size_t qo = base + (size_t)h * D + i;
  float x1 = qkv[qo], x2 = qkv[qo + half];
  qkv[qo]        =  x1 * c + x2 * s;
  qkv[qo + half] = -x1 * s + x2 * c;

  size_t ko = base + (size_t)C + (size_t)h * D + i;
  x1 = qkv[ko]; x2 = qkv[ko + half];
  qkv[ko]        =  x1 * c + x2 * s;
  qkv[ko + half] = -x1 * s + x2 * c;
}

// -------------------------------------------- sliding-window online softmax
// One thread per (b,h,t). <=65 keys; K/V rows L2-resident and shared between
// adjacent threads. Prefetches next K/V rows. Context written f16 [B*T, C].
__global__ void swa_kernel(const float* __restrict__ qkv,
                           half_t* __restrict__ ctx,
                           int Bb, int Tt, int C, int H, int D, int W) {
  int idx = blockIdx.x * blockDim.x + threadIdx.x;
  if (idx >= Bb * H * Tt) return;
  int t = idx % Tt;
  int h = (idx / Tt) % H;
  int b = idx / (Tt * H);
  const int C3 = 3 * C;

  const float* Qp = qkv + (size_t)(b * Tt + t) * C3 + h * D;
  float q[64];
#pragma unroll
  for (int d = 0; d < 64; ++d) q[d] = Qp[d];

  const float scale = rsqrtf((float)D);
  float m = -3.402823466e38f, l = 0.0f;
  float acc[64];
#pragma unroll
  for (int d = 0; d < 64; ++d) acc[d] = 0.0f;

  int k0 = t - W; if (k0 < 0) k0 = 0;
  for (int kk = k0; kk <= t; ++kk) {
    const float* Kp = qkv + (size_t)(b * Tt + kk) * C3 + C + h * D;
    __builtin_prefetch(Kp + C3, 0, 1);        // next key row
    __builtin_prefetch(Kp + C3 + C, 0, 1);    // next value row
    float s = 0.0f;
#pragma unroll
    for (int d = 0; d < 64; ++d) s = fmaf(q[d], Kp[d], s);
    s *= scale;
    float mn   = fmaxf(m, s);
    float corr = __expf(m - mn);   // 0 on the first iteration
    float p    = __expf(s - mn);
    l = l * corr + p;
    const float* Vp = Kp + C;
#pragma unroll
    for (int d = 0; d < 64; ++d) acc[d] = acc[d] * corr + p * Vp[d];
    m = mn;
  }
  float inv_l = 1.0f / l;
  half_t* Op = ctx + (size_t)(b * Tt + t) * C + h * D;
#pragma unroll
  for (int d = 0; d < 64; ++d) Op[d] = (half_t)(acc[d] * inv_l);
}

// ---------------------------------------------------------------------------
extern "C" void kernel_launch(void* const* d_in, const int* in_sizes, int n_in,
                              void* d_out, int out_size, void* d_ws, size_t ws_size,
                              hipStream_t stream) {
  const int B = 2, T = 2048, C = 1024, H = 16, D = 64, W = 64;
  const int M  = B * T;      // 4096 rows
  const int C3 = 3 * C;      // 3072

  const float* x      = (const float*)d_in[0];
  const float* w_attn = (const float*)d_in[1];
  const float* w_proj = (const float*)d_in[2];
  float* out = (float*)d_out;

  // workspace carve-up (72 MB; all 16B aligned)
  char* ws = (char*)d_ws;
  half_t* Xh   = (half_t*)ws;                                  // M*C
  half_t* WAh  = Xh  + (size_t)M  * C;                         // C3*C
  half_t* WPh  = WAh + (size_t)C3 * C;                         // C*C
  float*  qkv  = (float*)(WPh + (size_t)C * C);                // M*C3
  half_t* CTXh = (half_t*)(qkv + (size_t)M * C3);              // M*C

  // 1) casts to f16
  cast_f32_to_f16<<<(M * C + 255) / 256, 256, 0, stream>>>(x, Xh, M * C);
  cast_f32_to_f16<<<(C3 * C + 255) / 256, 256, 0, stream>>>(w_attn, WAh, C3 * C);
  cast_f32_to_f16<<<(C * C + 255) / 256, 256, 0, stream>>>(w_proj, WPh, C * C);

  // 2) qkv = x @ w_attn^T   (WMMA, async-to-LDS double buffered)
  dim3 g1(M / 64, C3 / 64);
  wmma_gemm_nt<<<g1, 128, 0, stream>>>(Xh, WAh, qkv, M, C3, C);

  // 3) RoPE on q,k (in place)
  int nrope = B * T * H * (D / 2);
  rope_inplace<<<(nrope + 255) / 256, 256, 0, stream>>>(qkv, B, T, C, H, D);

  // 4) sliding-window attention -> f16 context
  int natt = B * H * T;
  swa_kernel<<<(natt + 255) / 256, 256, 0, stream>>>(qkv, CTXh, B, T, C, H, D, W);

  // 5) y = ctx @ w_proj^T   (WMMA)
  dim3 g2(M / 64, C / 64);
  wmma_gemm_nt<<<g2, 128, 0, stream>>>(CTXh, WPh, out, M, C, C);
}
